// RPN_64622077936269
// MI455X (gfx1250) — matrix operand order repaired and checked by reading, not compiled, not run on hardware
//
#include <hip/hip_runtime.h>

// ---------------------------------------------------------------------------
// RPN head for MI455X (gfx1250): 3x3 conv (512->512, relu) via bf16 split-WMMA,
// then 1x1 cls/reg heads (TDM-staged) + sigmoid/threshold mask epilogue.
// ---------------------------------------------------------------------------

typedef __attribute__((ext_vector_type(16))) __bf16 v16bf;
typedef __attribute__((ext_vector_type(8)))  __bf16 v8bf;
typedef __attribute__((ext_vector_type(8)))  float  v8f;
typedef __attribute__((ext_vector_type(4)))  unsigned int v4u;
typedef __attribute__((ext_vector_type(8)))  int v8i;
typedef __attribute__((ext_vector_type(4)))  int v4i;

#define Bq   8
#define Hq   64
#define Wq   64
#define Cq   512
#define NCLS 9
#define NREG 36
#define NHEAD 45                       // 9 cls + 36 reg
static const size_t WT_ELEMS = (size_t)9 * Cq * Cq;   // [tap][n][k]

// ---------------------------------------------------------------------------
// Pack: W_base [3][3][cin][cout] -> WT_{hi,lo}[tap][n=cout][k=cin] bf16 split;
//       heads  -> WTH[45][512] fp32 (row-major per output channel).
// ---------------------------------------------------------------------------
__global__ __launch_bounds__(256) void rpn_pack(
    const float* __restrict__ W_base, const float* __restrict__ W_cls,
    const float* __restrict__ W_reg,
    unsigned short* __restrict__ wt_hi_u, unsigned short* __restrict__ wt_lo_u,
    float* __restrict__ wth)
{
  __bf16* wt_hi = (__bf16*)wt_hi_u;
  __bf16* wt_lo = (__bf16*)wt_lo_u;
  size_t gid = (size_t)blockIdx.x * 256 + threadIdx.x;
  if (gid < WT_ELEMS) {
    size_t k  = gid & 511;
    size_t tn = gid >> 9;
    size_t n  = tn & 511;
    size_t t  = tn >> 9;
    float v  = W_base[(t * Cq + k) * Cq + n];
    __bf16 h = (__bf16)v;
    wt_hi[gid] = h;
    wt_lo[gid] = (__bf16)(v - (float)h);
  } else if (gid < WT_ELEMS + (size_t)NHEAD * Cq) {
    size_t r  = gid - WT_ELEMS;
    size_t k  = r & 511;
    size_t oc = r >> 9;
    wth[r] = (oc < NCLS) ? W_cls[k * NCLS + oc] : W_reg[k * NREG + (oc - NCLS)];
  }
}

// Build a 16-element bf16 fragment from two 8-element (16B) runs.
__device__ inline v16bf make_frag(const __bf16* p0, const __bf16* p1)
{
  v8bf a = *(const v8bf*)p0;
  v8bf b = *(const v8bf*)p1;
  v16bf r;
#pragma unroll
  for (int i = 0; i < 8; ++i) { r[i] = a[i]; r[i + 8] = b[i]; }
  return r;
}

// ---------------------------------------------------------------------------
// 3x3 conv, SAME, relu.  Grid: 2048 WGs = (8 batch * 64 rows) * 4 N-quarters.
// WG = 256 threads = 8 wave32; each wave owns 4 16x16 fp32 accum tiles.
// K loop: 9 taps * 16 chunks of 32.  A staged through LDS as bf16 hi/lo.
// ---------------------------------------------------------------------------
__global__ __launch_bounds__(256) void rpn_conv3x3_wmma(
    const float* __restrict__ x,
    const unsigned short* __restrict__ wt_hi_u,
    const unsigned short* __restrict__ wt_lo_u,
    const float* __restrict__ b_base,
    float* __restrict__ feat)
{
  const __bf16* wt_hi = (const __bf16*)wt_hi_u;
  const __bf16* wt_lo = (const __bf16*)wt_lo_u;

  __shared__ __bf16 As_hi[Wq * 32];   // [pixel][k] 4 KiB
  __shared__ __bf16 As_lo[Wq * 32];   // 4 KiB

  const int wgid = blockIdx.x;
  const int nblk = wgid & 3;          // which 128-wide N quarter
  const int row  = wgid >> 2;
  const int y    = row & 63;
  const int b    = row >> 6;

  const int tid   = threadIdx.x;
  const int lane  = tid & 31;
  const int wave  = tid >> 5;
  const int mbase = (wave & 3) * 16;                 // pixel tile base
  const int nbaseW = nblk * 128 + (wave >> 2) * 64;  // channel tile base

  const int m = lane & 15;      // row-in-tile for A, col-in-tile for B
  const int g = lane >> 4;      // lane half

  v8f acc[4];
#pragma unroll
  for (int i = 0; i < 4; ++i)
#pragma unroll
    for (int j = 0; j < 8; ++j) acc[i][j] = 0.0f;

  const int sp = tid >> 2;            // staging pixel 0..63
  const int sk = (tid & 3) * 8;       // staging k offset 0..24

  for (int t = 0; t < 9; ++t) {
    const int dy = t / 3 - 1;
    const int dx = t % 3 - 1;
    const int ys = y + dy;
    if (ys < 0 || ys >= Hq) continue;          // uniform across WG

    for (int kb = 0; kb < Cq; kb += 32) {
      // ---- stage A chunk: 64 pixels x 32 k, fp32 -> bf16 hi/lo ----
      {
        const int xs = sp + dx;
        float4 v0 = make_float4(0.f, 0.f, 0.f, 0.f);
        float4 v1 = v0;
        if (xs >= 0 && xs < Wq) {
          const float4* src = (const float4*)(x +
              ((((size_t)b * Hq + ys) * Wq + xs) * Cq + kb + sk));
          v0 = src[0];
          v1 = src[1];
        }
        float vv[8] = {v0.x, v0.y, v0.z, v0.w, v1.x, v1.y, v1.z, v1.w};
#pragma unroll
        for (int j = 0; j < 8; ++j) {
          float f  = vv[j];
          __bf16 h = (__bf16)f;
          As_hi[sp * 32 + sk + j] = h;
          As_lo[sp * 32 + sk + j] = (__bf16)(f - (float)h);
        }
      }
      __syncthreads();

      // ---- per-wave fragments + WMMA ----
      const __bf16* arow_hi = &As_hi[(mbase + m) * 32];
      const __bf16* arow_lo = &As_lo[(mbase + m) * 32];
      v16bf ahi = make_frag(arow_hi + 8 * g, arow_hi + 16 + 8 * g);
      v16bf alo = make_frag(arow_lo + 8 * g, arow_lo + 16 + 8 * g);

#pragma unroll
      for (int nt = 0; nt < 4; ++nt) {
        const int n = nbaseW + nt * 16 + m;    // this lane's B column
        const size_t woff = ((size_t)t * Cq + n) * Cq + kb + 8 * g;
        v16bf bhi = make_frag(wt_hi + woff, wt_hi + woff + 16);
        v16bf blo = make_frag(wt_lo + woff, wt_lo + woff + 16);
        if (kb + 32 < Cq) {                    // prefetch next K-chunk of B
          __builtin_prefetch(wt_hi + woff + 32, 0, 1);
          __builtin_prefetch(wt_lo + woff + 32, 0, 1);
        }
        // acc += Ahi*Bhi + Ahi*Blo + Alo*Bhi   (split-bf16 ~fp32)
        acc[nt] = __builtin_amdgcn_wmma_f32_16x16x32_bf16(
            false, ahi, false, bhi, (short)0, acc[nt], false, false);
        acc[nt] = __builtin_amdgcn_wmma_f32_16x16x32_bf16(
            false, ahi, false, blo, (short)0, acc[nt], false, false);
        acc[nt] = __builtin_amdgcn_wmma_f32_16x16x32_bf16(
            false, alo, false, bhi, (short)0, acc[nt], false, false);
      }
      __syncthreads();
    }
  }

  // ---- epilogue: bias + relu, store fp32 feature map ----
#pragma unroll
  for (int nt = 0; nt < 4; ++nt) {
    const int n = nbaseW + nt * 16 + m;
    const float bias = b_base[n];
#pragma unroll
    for (int r = 0; r < 8; ++r) {
      const int pix = mbase + r + 8 * g;       // C/D layout: row = r + 8*laneHalf
      float v = acc[nt][r] + bias;
      v = v > 0.0f ? v : 0.0f;
      feat[(((size_t)b * Hq + y) * Wq + pix) * Cq + n] = v;
    }
  }
}

// ---------------------------------------------------------------------------
// Heads: per pixel, 45 dots over 512 channels; sigmoid + size mask epilogue.
// WG = 256 threads = 4 pixels * 64 threads.
// Feature rows are staged global->LDS by the Tensor Data Mover (one wave
// issues a 2D D#: 4 rows x 512 fp32, stride 512; LDS dest offset 0).
// ---------------------------------------------------------------------------
__global__ __launch_bounds__(256) void rpn_heads(
    const float* __restrict__ feat, const float* __restrict__ wth,
    const float* __restrict__ b_cls, const float* __restrict__ b_reg,
    float* __restrict__ out)
{
  extern __shared__ float smem_f[];            // dynamic LDS: base offset 0
  float* fbuf = smem_f;                        // [4][512] @ LDS offset 0
  float* res  = smem_f + 4 * Cq;               // [4][45]

  const int tid = threadIdx.x;
  const int pg  = tid >> 6;
  const int l   = tid & 63;
  const size_t pix = (size_t)blockIdx.x * 4 + pg;

  // ---- TDM: copy feat[blockIdx*4 .. +3][0..511] into LDS @ 0 ----
  if (tid < 32) {                              // wave 0 only (uniform branch)
    const unsigned long long gaddr =
        (unsigned long long)(feat + (size_t)blockIdx.x * 4 * Cq);
    v4u g0;
    g0[0] = 1u;                                              // count = 1
    g0[1] = 0u;                                              // lds_addr = 0
    g0[2] = (unsigned)(gaddr & 0xFFFFFFFFu);                 // global_addr[31:0]
    g0[3] = (unsigned)((gaddr >> 32) & 0x01FFFFFFu) | (2u << 30); // [56:32]+type=2
    v8i g1;
    g1[0] = (int)(2u << 16);                 // data_size = 2 -> 4-byte elements
    g1[1] = (int)(512u << 16);               // tensor_dim0[15:0] = 512
    g1[2] = (int)(4u << 16);                 // tensor_dim0[31:16]=0, tensor_dim1=4
    g1[3] = (int)(512u << 16);               // tensor_dim1 hi=0, tile_dim0 = 512
    g1[4] = 4;                               // tile_dim1 = 4, tile_dim2 = 0
    g1[5] = 512;                             // tensor_dim0_stride[31:0] = 512
    g1[6] = 0;                               // stride hi / tensor_dim1_stride lo
    g1[7] = 0;
    v4i g2 = {0, 0, 0, 0};
    v4i g3 = {0, 0, 0, 0};
    v8i g4 = {0, 0, 0, 0, 0, 0, 0, 0};       // unused extra group (6-arg form)
    __builtin_amdgcn_tensor_load_to_lds(g0, g1, g2, g3, g4, 0);
    __builtin_amdgcn_s_wait_tensorcnt(0);
  }
  __syncthreads();

  if (l < NHEAD) {
    const float* wrow = wth + (size_t)l * Cq;
    const float* frow = fbuf + pg * Cq;
    float s = 0.0f;
    for (int k = 0; k < Cq; k += 4) {
      float4 f = *(const float4*)&frow[k];
      float4 w = *(const float4*)&wrow[k];
      s += f.x * w.x + f.y * w.y + f.z * w.z + f.w * w.w;
    }
    s += (l < NCLS) ? b_cls[l] : b_reg[l - NCLS];
    res[pg * NHEAD + l] = s;
  }
  __syncthreads();

  if (l < NCLS) {
    const int a = l;
    const float* r = res + pg * NHEAD;
    const float obj = 1.0f / (1.0f + __expf(-r[a]));
    const float b0 = r[NCLS + a * 4 + 0];
    const float b1 = r[NCLS + a * 4 + 1];
    const float b2 = r[NCLS + a * 4 + 2];
    const float b3 = r[NCLS + a * 4 + 3];
    const float mm = ((obj > 0.7f) && (b2 > 10.0f) && (b3 > 10.0f)) ? 1.0f : 0.0f;
    float* o = out + (pix * NCLS + a) * 5;
    o[0] = b0 * mm;
    o[1] = b1 * mm;
    o[2] = b2 * mm;
    o[3] = b3 * mm;
    o[4] = obj * mm;
  }
}

// ---------------------------------------------------------------------------
// Launch.  Workspace layout (bytes):
//   [0)                 WT_hi  : 9*512*512 bf16 = 4,718,592
//   [4718592)           WT_lo  : 4,718,592
//   [9437184)           WTH    : 45*512 fp32    = 92,160
//   [9529344)           feat   : 8*64*64*512 fp32 = 67,108,864   (total ~73 MiB)
// ---------------------------------------------------------------------------
extern "C" void kernel_launch(void* const* d_in, const int* in_sizes, int n_in,
                              void* d_out, int out_size, void* d_ws, size_t ws_size,
                              hipStream_t stream)
{
  const float* x      = (const float*)d_in[0];
  const float* W_base = (const float*)d_in[1];
  const float* b_base = (const float*)d_in[2];
  const float* W_cls  = (const float*)d_in[3];
  const float* b_cls  = (const float*)d_in[4];
  const float* W_reg  = (const float*)d_in[5];
  const float* b_reg  = (const float*)d_in[6];
  float* out = (float*)d_out;

  char* ws = (char*)d_ws;
  unsigned short* wt_hi = (unsigned short*)(ws);
  unsigned short* wt_lo = (unsigned short*)(ws + WT_ELEMS * 2);
  float*          wth   = (float*)(ws + 2 * WT_ELEMS * 2);
  float*          feat  = (float*)(ws + 2 * WT_ELEMS * 2 + (size_t)NHEAD * Cq * 4);

  {
    const size_t total = WT_ELEMS + (size_t)NHEAD * Cq;   // 2,382,336
    const int blocks = (int)((total + 255) / 256);
    rpn_pack<<<blocks, 256, 0, stream>>>(W_base, W_cls, W_reg, wt_hi, wt_lo, wth);
  }
  {
    const int blocks = Bq * Hq * 4;                       // 2048
    rpn_conv3x3_wmma<<<blocks, 256, 0, stream>>>(x, wt_hi, wt_lo, b_base, feat);
  }
  {
    const int blocks = (Bq * Hq * Wq) / 4;                // 8192
    const size_t shmem = (size_t)(4 * Cq + 4 * NHEAD) * sizeof(float); // 8912 B
    rpn_heads<<<blocks, 256, shmem, stream>>>(feat, wth, b_cls, b_reg, out);
  }
}
